// Transformer_2147483648333
// MI455X (gfx1250) — matrix operand order repaired and checked by reading, not compiled, not run on hardware
//
#include <hip/hip_runtime.h>

// ---------------------------------------------------------------------------
// MI455X (gfx1250) fused QKV + flash-attention, bf16 WMMA path + TDM staging.
//   B=4, S=4096, D_IN=256, UNITS=64
// Kernel 1: qkv_proj_wmma  -> Q(*0.125*log2e), K row-major, V transposed (bf16, d_ws)
// Kernel 2: flash_attn_wmma -> K blocks staged in LDS via tensor_load_to_lds
//           (double-buffered, TENSORcnt), online softmax (exp2), f32 out
// ---------------------------------------------------------------------------

typedef __bf16 bf16_t;
typedef __attribute__((ext_vector_type(16))) __bf16 v16bf;
typedef __attribute__((ext_vector_type(8)))  float  v8f;
typedef __attribute__((ext_vector_type(4)))  unsigned int v4u;
typedef __attribute__((ext_vector_type(8)))  int  v8i;
typedef __attribute__((ext_vector_type(4)))  int  v4i;

union V16U { v16bf v; uint4 q[2]; };

// f32 -> bf16 native conversion (lowers to v_cvt_pk_bf16_f32 on gfx1250)
__device__ __forceinline__ bf16_t f2bf(float f) { return (bf16_t)f; }

#define WMMA_BF16(A, Bm, C) \
    __builtin_amdgcn_wmma_f32_16x16x32_bf16(false, (A), false, (Bm), (short)0, (C), false, false)

// 1/sqrt(64) * log2(e): scores leave QK^T already in log2 domain,
// so softmax uses raw v_exp_f32 (exp2) with no per-element multiply.
#define QSCALE 0.1803368801111204f

// K-tile LDS row stride: 64 bf16 payload + 8 bf16 TDM pad = 72 elements
// (36 dwords; lane stride 4*(9n mod 16) mod 64 banks -> conflict-free b128)
#define KSTR 72

// ---------------------------------------------------------------------------
// TDM: DMA one 128x64 bf16 K block (row-major, contiguous) into LDS with
// 4-dword padding after each 32-dword row. D# packed per CDNA5 ISA Ch.10.8.
// This toolchain exposes the 6-arg builtin:
//   (uint32x4 g0, int32x8 g1, int32x4 g2, int32x4 g3, int32x8 gx, i32 cpol)
// ---------------------------------------------------------------------------
__device__ __forceinline__ void tdm_load_k_block(const bf16_t* gptr, unsigned lds_byte_off) {
    const unsigned long long ga = (unsigned long long)(const void*)gptr;
    v4u g0;
    g0[0] = 1u;                                        // count=1, user mode
    g0[1] = lds_byte_off;                              // lds_addr
    g0[2] = (unsigned)(ga & 0xFFFFFFFFu);              // global_addr[31:0]
    g0[3] = (unsigned)((ga >> 32) & 0x01FFFFFFu)       // global_addr[56:32]
            | 0x80000000u;                             // type=2 ("image")
    v8i g1;
    g1[0] = (int)((1u << 16)     // data_size = 1 -> 2 bytes
                | (1u << 20)     // pad_enable
                | (4u << 22)     // pad_interval = 32 dwords (one 64-elem row)
                | (3u << 25));   // pad_amount   = 4 dwords (8 bf16)
    g1[1] = (int)(64u << 16);    // [31:16] tensor_dim0 lo = 64
    g1[2] = (int)(128u << 16);   // [15:0] dim0 hi = 0, [31:16] tensor_dim1 lo = 128
    g1[3] = (int)(64u << 16);    // [15:0] dim1 hi = 0, [31:16] tile_dim0 = 64
    g1[4] = (int)(128u);         // [15:0] tile_dim1 = 128, [31:16] tile_dim2 = 0
    g1[5] = 64;                  // tensor_dim0_stride[31:0] = 64 elements
    g1[6] = 0;                   // stride0 hi / stride1 lo
    g1[7] = 0;                   // stride1 hi
    const v4i gz4 = (v4i){0, 0, 0, 0};                 // 2-D: groups 2/3 unused
    const v8i gz8 = (v8i){0, 0, 0, 0, 0, 0, 0, 0};
    __builtin_amdgcn_tensor_load_to_lds(g0, g1, gz4, gz4, gz8, 0);
}

// ---------------------------------------------------------------------------
// Kernel 1: Q/K/V projection.
// Grid: 128 blocks x 256 threads (8 waves). Each block: 128 rows of x.
// Each wave owns one 16-row tile and all 12 (matrix, col-tile) outputs.
// Weights staged transposed in LDS: Wt[mat][u][d], stride 264 bf16
// (132 dwords; 132 mod 64 = 4 -> lane n hits banks 4n..4n+3: conflict-free).
// ---------------------------------------------------------------------------
__global__ void __launch_bounds__(256)
qkv_proj_wmma(const float* __restrict__ x,
              const float* __restrict__ Wq, const float* __restrict__ bq,
              const float* __restrict__ Wk, const float* __restrict__ bk,
              const float* __restrict__ Wv, const float* __restrict__ bv,
              bf16_t* __restrict__ qg, bf16_t* __restrict__ kg,
              bf16_t* __restrict__ vt)
{
    constexpr int DIN = 256, U = 64, WSTR = 264;
    extern __shared__ bf16_t lds[];            // [3][64][264] bf16 = 101376 B

    const int tid = threadIdx.x;

    const float* Ws[3] = { Wq, Wk, Wv };
#pragma unroll
    for (int m = 0; m < 3; ++m) {
        const float* W = Ws[m];
        bf16_t* dst = lds + m * (U * WSTR);
        for (int idx = tid; idx < DIN * U; idx += 256) {
            const int d = idx >> 6;
            const int u = idx & 63;
            dst[u * WSTR + d] = f2bf(W[idx]);
        }
    }
    __syncthreads();

    const int wave = tid >> 5;
    const int lane = tid & 31;
    const int lh   = lane >> 4;
    const int ln   = lane & 15;
    const long rbase = (long)blockIdx.x * 128 + wave * 16;

    v8f acc[3][4];
#pragma unroll
    for (int m = 0; m < 3; ++m)
#pragma unroll
        for (int ct = 0; ct < 4; ++ct)
            acc[m][ct] = (v8f){};

#pragma unroll
    for (int kk = 0; kk < 8; ++kk) {
        // A-frag (16x32): half0 K = kk*32+{0..7,16..23}; half1 {8..15,24..31}
        const float* xrow = x + (rbase + ln) * DIN + kk * 32 + lh * 8;
        const float4 f0 = *(const float4*)(xrow + 0);
        const float4 f1 = *(const float4*)(xrow + 4);
        const float4 f2 = *(const float4*)(xrow + 16);
        const float4 f3 = *(const float4*)(xrow + 20);
        v16bf a;
        a[0]  = f2bf(f0.x); a[1]  = f2bf(f0.y); a[2]  = f2bf(f0.z); a[3]  = f2bf(f0.w);
        a[4]  = f2bf(f1.x); a[5]  = f2bf(f1.y); a[6]  = f2bf(f1.z); a[7]  = f2bf(f1.w);
        a[8]  = f2bf(f2.x); a[9]  = f2bf(f2.y); a[10] = f2bf(f2.z); a[11] = f2bf(f2.w);
        a[12] = f2bf(f3.x); a[13] = f2bf(f3.y); a[14] = f2bf(f3.z); a[15] = f2bf(f3.w);

#pragma unroll
        for (int m = 0; m < 3; ++m) {
#pragma unroll
            for (int ct = 0; ct < 4; ++ct) {
                const bf16_t* wp = lds + m * (U * WSTR) + (ct * 16 + ln) * WSTR
                                   + kk * 32 + lh * 8;
                V16U bu;
                bu.q[0] = *(const uint4*)(wp + 0);
                bu.q[1] = *(const uint4*)(wp + 16);
                acc[m][ct] = WMMA_BF16(a, bu.v, acc[m][ct]);
            }
        }
    }

    const float* bs[3] = { bq, bk, bv };
#pragma unroll
    for (int m = 0; m < 3; ++m) {
#pragma unroll
        for (int ct = 0; ct < 4; ++ct) {
            const int colu = ct * 16 + ln;
            const float bias = bs[m][colu];
#pragma unroll
            for (int r = 0; r < 8; ++r) {
                const long row = rbase + r + lh * 8;
                const float val = acc[m][ct][r] + bias;
                if (m == 0) {
                    qg[row * 64 + colu] = f2bf(val * QSCALE);
                } else if (m == 1) {
                    kg[row * 64 + colu] = f2bf(val);
                } else {
                    const long b = row >> 12;
                    const long s = row & 4095;
                    vt[(b * 64 + colu) * 4096 + s] = f2bf(val);
                }
            }
        }
    }
}

// ---------------------------------------------------------------------------
// Kernel 2: flash attention. Grid: 128 blocks (4 batches x 32 q-blocks),
// 256 threads (8 waves), each wave owns 16 query rows, 64-col f32 accumulator.
// K blocks arrive in LDS via double-buffered TDM; V read from global (transposed).
// LDS layout: [2][128][72] bf16 K buffers (36864 B) then 8x[16][136] P staging.
// ---------------------------------------------------------------------------
__global__ void __launch_bounds__(256)
flash_attn_wmma(const bf16_t* __restrict__ qg, const bf16_t* __restrict__ kg,
                const bf16_t* __restrict__ vt, float* __restrict__ out)
{
    constexpr int S = 4096, PSTR = 136;
    constexpr int KBUF = 128 * KSTR;           // elements per K buffer
    extern __shared__ bf16_t plds[];           // 36864 + 34816 = 71680 B

    const int tid  = threadIdx.x;
    const int wave = tid >> 5;
    const int lane = tid & 31;
    const int lh   = lane >> 4;
    const int ln   = lane & 15;
    const int b    = blockIdx.x >> 5;          // 32 q-blocks per batch
    const int qblk = blockIdx.x & 31;
    const long qrow = (long)b * S + qblk * 128 + wave * 16;
    bf16_t* kbuf = plds;                       // [2][128][KSTR]
    bf16_t* pbuf = plds + 2 * KBUF + wave * (16 * PSTR);
    const unsigned kbuf_off =
        (unsigned)((unsigned long long)(const void*)kbuf & 0xFFFFFFFFu);

    // Q A-frags: 2 u-steps of K=32 (UNITS=64), loaded once.
    v16bf qa[2];
#pragma unroll
    for (int us = 0; us < 2; ++us) {
        const bf16_t* qp = qg + (qrow + ln) * 64 + us * 32 + lh * 8;
        V16U t;
        t.q[0] = *(const uint4*)(qp + 0);
        t.q[1] = *(const uint4*)(qp + 16);
        qa[us] = t.v;
    }

    v8f accv[4];
#pragma unroll
    for (int ut = 0; ut < 4; ++ut) accv[ut] = (v8f){};
    float mrow[8], lrow[8];
#pragma unroll
    for (int r = 0; r < 8; ++r) { mrow[r] = -__builtin_inff(); lrow[r] = 0.0f; }

    // Prologue: stage key block 0
    if (wave == 0)
        tdm_load_k_block(kg + ((long)b * S) * 64, kbuf_off);

    for (int kb = 0; kb < 32; ++kb) {
        if (wave == 0)
            __builtin_amdgcn_s_wait_tensorcnt(0);   // block kb landed in LDS
        __syncthreads();                            // visible to all waves; also
                                                    // guards buffer reuse (see below)
        // Issue DMA for block kb+1 into the other buffer. The barrier above is
        // after every wave finished reading that buffer in iteration kb-1.
        if (wave == 0 && kb < 31)
            tdm_load_k_block(kg + ((long)b * S + (long)(kb + 1) * 128) * 64,
                             kbuf_off + (unsigned)(((kb + 1) & 1) * KBUF * 2));

        const bf16_t* kblk = kbuf + (kb & 1) * KBUF;

        // Prefetch next V lines (global_prefetch_b8)
        if (kb < 31)
            __builtin_prefetch(vt + ((long)b * 64 + (lane << 1)) * 4096
                               + (kb + 1) * 128, 0, 0);

        // ---- S = Q K^T (log2 domain): 8 n-tiles x 2 u-steps, K from LDS
        v8f st[8];
#pragma unroll
        for (int j = 0; j < 8; ++j) {
            v8f c = (v8f){};
#pragma unroll
            for (int us = 0; us < 2; ++us) {
                const bf16_t* kp = kblk + (j * 16 + ln) * KSTR + us * 32 + lh * 8;
                V16U bu;
                bu.q[0] = *(const uint4*)(kp + 0);
                bu.q[1] = *(const uint4*)(kp + 16);
                c = WMMA_BF16(qa[us], bu.v, c);
            }
            st[j] = c;
        }

        // ---- online softmax row stats (rows live in VGPR index r, halves)
        float mnew[8], fscl[8], rsum[8];
#pragma unroll
        for (int r = 0; r < 8; ++r) {
            float v = st[0][r];
#pragma unroll
            for (int j = 1; j < 8; ++j) v = fmaxf(v, st[j][r]);
            v = fmaxf(v, __shfl_xor(v, 1, 32));
            v = fmaxf(v, __shfl_xor(v, 2, 32));
            v = fmaxf(v, __shfl_xor(v, 4, 32));
            v = fmaxf(v, __shfl_xor(v, 8, 32));
            mnew[r] = fmaxf(mrow[r], v);
            fscl[r] = exp2f(mrow[r] - mnew[r]);   // 0 on first block (-inf)
            rsum[r] = 0.0f;
        }

        // ---- P = exp2(S - m), stage in own-wave LDS region
#pragma unroll
        for (int j = 0; j < 8; ++j) {
#pragma unroll
            for (int r = 0; r < 8; ++r) {
                const float p = exp2f(st[j][r] - mnew[r]);
                rsum[r] += p;
                pbuf[(r + lh * 8) * PSTR + j * 16 + ln] = f2bf(p);
            }
        }

#pragma unroll
        for (int r = 0; r < 8; ++r) {
            float v = rsum[r];
            v += __shfl_xor(v, 1, 32);
            v += __shfl_xor(v, 2, 32);
            v += __shfl_xor(v, 4, 32);
            v += __shfl_xor(v, 8, 32);
            lrow[r] = lrow[r] * fscl[r] + v;
            mrow[r] = mnew[r];
        }
#pragma unroll
        for (int ut = 0; ut < 4; ++ut)
#pragma unroll
            for (int r = 0; r < 8; ++r)
                accv[ut][r] *= fscl[r];

        // ---- acc += P V : 4 t-steps (K=32 keys) x 4 u-tiles
        // (same-wave DS ops are in-order; loads below see the stores above)
#pragma unroll
        for (int t = 0; t < 4; ++t) {
            const bf16_t* pp = pbuf + ln * PSTR + t * 32 + lh * 8;
            V16U a;
            a.q[0] = *(const uint4*)(pp + 0);
            a.q[1] = *(const uint4*)(pp + 16);
#pragma unroll
            for (int ut = 0; ut < 4; ++ut) {
                const bf16_t* vp = vt + ((long)b * 64 + ut * 16 + ln) * 4096
                                   + kb * 128 + t * 32 + lh * 8;
                V16U bu;
                bu.q[0] = *(const uint4*)(vp + 0);
                bu.q[1] = *(const uint4*)(vp + 16);
                accv[ut] = WMMA_BF16(a.v, bu.v, accv[ut]);
            }
        }
    }

    // ---- normalize and store f32 output [B, S, 64]
#pragma unroll
    for (int ut = 0; ut < 4; ++ut) {
#pragma unroll
        for (int r = 0; r < 8; ++r) {
            const long row = qrow + r + lh * 8;
            out[row * 64 + ut * 16 + ln] = accv[ut][r] / lrow[r];
        }
    }
}

// ---------------------------------------------------------------------------
extern "C" void kernel_launch(void* const* d_in, const int* in_sizes, int n_in,
                              void* d_out, int out_size, void* d_ws, size_t ws_size,
                              hipStream_t stream) {
    (void)in_sizes; (void)n_in; (void)out_size; (void)ws_size;
    const float* x  = (const float*)d_in[0];
    const float* Wq = (const float*)d_in[1];
    const float* bq = (const float*)d_in[2];
    const float* Wk = (const float*)d_in[3];
    const float* bk = (const float*)d_in[4];
    const float* Wv = (const float*)d_in[5];
    const float* bv = (const float*)d_in[6];
    float* out = (float*)d_out;

    // Workspace: Q, K, Vt as bf16, 4*4096*64 = 1048576 elements each (6 MB)
    bf16_t* qws = (bf16_t*)d_ws;
    bf16_t* kws = qws + 1048576;
    bf16_t* vws = kws + 1048576;

    constexpr size_t LDS1 = 3 * 64 * 264 * sizeof(bf16_t);              // 101376 B
    constexpr size_t LDS2 = (2 * 128 * KSTR + 8 * 16 * 136) * sizeof(bf16_t); // 71680 B

    qkv_proj_wmma<<<128, 256, LDS1, stream>>>(x, Wq, bq, Wk, bk, Wv, bv,
                                              qws, kws, vws);
    flash_attn_wmma<<<128, 256, LDS2, stream>>>(qws, kws, vws, out);
}
